// GATBaseline_51694226375452
// MI455X (gfx1250) — matrix operand order repaired and checked by reading, not compiled, not run on hardware
//
#include <hip/hip_runtime.h>
#include <hip/hip_bf16.h>
#include <math.h>

typedef __attribute__((ext_vector_type(16))) _Float16 v16h;
typedef __attribute__((ext_vector_type(8)))  _Float16 v8h;
typedef __attribute__((ext_vector_type(8)))  float    v8f;

#define TPB 256
#define NEG_SLOPE 0.2f

// ---- order-preserving float <-> uint map (for exact atomic segment-max) ----
__device__ __forceinline__ unsigned f2ord(float f) {
  unsigned u = __float_as_uint(f);
  return (u & 0x80000000u) ? ~u : (u | 0x80000000u);
}
__device__ __forceinline__ float ord2f(unsigned u) {
  return (u & 0x80000000u) ? __uint_as_float(u & 0x7fffffffu) : __uint_as_float(~u);
}

// ---------------- utility kernels ----------------
__global__ void fill_u32(unsigned* __restrict__ p, unsigned v, int n) {
  int i = blockIdx.x * blockDim.x + threadIdx.x;
  if (i < n) p[i] = v;
}

__global__ void f32_to_f16(const float* __restrict__ in, _Float16* __restrict__ out, int n) {
  int i = blockIdx.x * blockDim.x + threadIdx.x;
  if (i < n) out[i] = (_Float16)in[i];
}

// W[K,M] f32 -> Wt[M,K] f16 (so WMMA B columns are contiguous)
__global__ void transpose_to_f16(const float* __restrict__ W, _Float16* __restrict__ Wt,
                                 int K, int M) {
  int i = blockIdx.x * blockDim.x + threadIdx.x;
  if (i < K * M) {
    int k = i / M, m = i - k * M;
    Wt[(size_t)m * K + k] = (_Float16)W[i];
  }
}

// ---------------- WMMA GEMM: Out[N,M] = A_f16[N,K] @ Wt_f16[M,K]^T ----------------
// One wave per 16x16 output tile, K stepped by 32 (v_wmma_f32_16x16x32_f16).
__global__ void wmma_gemm_f16(const _Float16* __restrict__ A, const _Float16* __restrict__ Wt,
                              float* __restrict__ Out, int Nrows, int K, int M) {
  int wid  = (int)((blockIdx.x * blockDim.x + threadIdx.x) >> 5);
  int lane = threadIdx.x & 31;
  int tilesN = M >> 4;
  int tm = wid / tilesN;
  int tn = wid - tm * tilesN;
  if (tm * 16 >= Nrows) return;           // wave-uniform guard
  int half = lane >> 4;                    // 0: lanes 0-15, 1: lanes 16-31
  int l    = lane & 15;

  const _Float16* arow = A  + (size_t)(tm * 16 + l) * K;   // A row M=l
  const _Float16* bcol = Wt + (size_t)(tn * 16 + l) * K;   // B column N=l

  v8f c = {};
  for (int k0 = 0; k0 < K; k0 += 32) {
    // A 16-bit layout: lanes(half=0): K = k0+[0..7] and k0+16+[0..7]; half=1: +8
    v8h alo = *(const v8h*)(arow + k0 + half * 8);
    v8h ahi = *(const v8h*)(arow + k0 + half * 8 + 16);
    // B 16-bit layout: lane=N, K = k0+half*8+[0..7] and k0+16+half*8+[0..7] (same as A)
    const v8h* pb = (const v8h*)(bcol + k0 + half * 8);
    v8h blo = pb[0];
    v8h bhi = pb[2];
    v16h av, bv;
#pragma unroll
    for (int i = 0; i < 8; ++i) {
      av[i] = alo[i]; av[i + 8] = ahi[i];
      bv[i] = blo[i]; bv[i + 8] = bhi[i];
    }
    c = __builtin_amdgcn_wmma_f32_16x16x32_f16(false, av, false, bv,
                                               (short)0, c, false, false);
  }
  // C/D layout: VGPR j -> row (tm*16 + half*8 + j), col (tn*16 + l)
  float* o = Out + (size_t)(tm * 16 + half * 8) * M + tn * 16 + l;
#pragma unroll
  for (int j = 0; j < 8; ++j) o[(size_t)j * M] = c[j];
}

// ---------------- attention pieces ----------------
// e_src[n,h] = sum_c h[n,h,c]*a_src[h,c]; e_dst likewise
__global__ void node_scores(const float* __restrict__ h, const float* __restrict__ asrc,
                            const float* __restrict__ adst, float* __restrict__ esrc,
                            float* __restrict__ edst, int Nn, int Hh, int C) {
  int i = blockIdx.x * blockDim.x + threadIdx.x;
  if (i >= Nn * Hh) return;
  int n = i / Hh, hh = i - n * Hh;
  const float* row = h + (size_t)n * Hh * C + (size_t)hh * C;
  const float* as  = asrc + hh * C;
  const float* ad  = adst + hh * C;
  float s0 = 0.f, s1 = 0.f;
#pragma unroll 4
  for (int cc = 0; cc < C; ++cc) { float v = row[cc]; s0 += v * as[cc]; s1 += v * ad[cc]; }
  esrc[i] = s0;
  edst[i] = s1;
}

// e = leaky_relu(e_src[src]+e_dst[dst]); atomic segment-max over dst (ordered-uint)
__global__ void edge_scores(const int* __restrict__ ei, int E, int Etot, int Hh,
                            const float* __restrict__ esrc, const float* __restrict__ edst,
                            float* __restrict__ ebuf, unsigned* __restrict__ mbuf) {
  int idx = blockIdx.x * blockDim.x + threadIdx.x;
  if (idx >= Etot * Hh) return;
  int e = idx / Hh, hh = idx - e * Hh;
  int s = (e < E) ? ei[e]     : (e - E);   // self-loops appended
  int d = (e < E) ? ei[E + e] : (e - E);
  float v = esrc[(size_t)s * Hh + hh] + edst[(size_t)d * Hh + hh];
  v = (v > 0.f) ? v : NEG_SLOPE * v;
  ebuf[idx] = v;
  atomicMax(&mbuf[(size_t)d * Hh + hh], f2ord(v));
}

// decode ordered-uint max back to float in place (init sentinel -> 0, like reference)
__global__ void decode_m(unsigned* __restrict__ m, int n, unsigned initv) {
  int i = blockIdx.x * blockDim.x + threadIdx.x;
  if (i < n) {
    unsigned u = m[i];
    ((float*)m)[i] = (u == initv) ? 0.f : ord2f(u);
  }
}

// ex = exp(e - m[dst]); atomic segment-sum into denom
__global__ void edge_expsum(const int* __restrict__ ei, int E, int Etot, int Hh,
                            const float* __restrict__ mflt, float* __restrict__ ebuf,
                            float* __restrict__ denom) {
  int idx = blockIdx.x * blockDim.x + threadIdx.x;
  if (idx >= Etot * Hh) return;
  int e = idx / Hh, hh = idx - e * Hh;
  int d = (e < E) ? ei[E + e] : (e - E);
  float ex = __expf(ebuf[idx] - mflt[(size_t)d * Hh + hh]);
  ebuf[idx] = ex;
  atomicAdd(&denom[(size_t)d * Hh + hh], ex);
}

// out[dst,h,c] += h[src,h,c] * ex/(denom[dst,h]+eps)   (L2-resident atomics)
__global__ void edge_aggregate(const int* __restrict__ ei, int E, int Etot, int Hh, int C,
                               const float* __restrict__ hPre, const float* __restrict__ ebuf,
                               const float* __restrict__ denom, float* __restrict__ out) {
  int M = Hh * C;
  long long idx = (long long)blockIdx.x * blockDim.x + threadIdx.x;
  long long total = (long long)Etot * M;
  if (idx >= total) return;
  int e  = (int)(idx / M);
  int j  = (int)(idx - (long long)e * M);
  int hh = j / C;
  int s = (e < E) ? ei[e]     : (e - E);
  int d = (e < E) ? ei[E + e] : (e - E);
  float alpha = ebuf[(size_t)e * Hh + hh] / (denom[(size_t)d * Hh + hh] + 1e-16f);
  atomicAdd(&out[(size_t)d * M + j], hPre[(size_t)s * M + j] * alpha);
}

__global__ void bias_act(float* __restrict__ out, const float* __restrict__ b,
                         int total, int M, int relu) {
  int i = blockIdx.x * blockDim.x + threadIdx.x;
  if (i >= total) return;
  float v = out[i] + b[i % M];
  if (relu) v = fmaxf(v, 0.f);
  out[i] = v;
}

// ---------------- pooling + head ----------------
__global__ void pool_counts(const int* __restrict__ batch, float* __restrict__ counts, int Nn) {
  int n = blockIdx.x * blockDim.x + threadIdx.x;
  if (n < Nn) atomicAdd(&counts[batch[n]], 1.0f);
}
__global__ void pool_sum(const float* __restrict__ h, const int* __restrict__ batch,
                         float* __restrict__ pooled, int Nn, int C) {
  int i = blockIdx.x * blockDim.x + threadIdx.x;
  if (i >= Nn * C) return;
  int n = i / C, cc = i - n * C;
  atomicAdd(&pooled[(size_t)batch[n] * C + cc], h[i]);
}
__global__ void final_linear(const float* __restrict__ pooled, const float* __restrict__ counts,
                             const float* __restrict__ lw, const float* __restrict__ lb,
                             float* __restrict__ out, int G, int C) {
  int g = blockIdx.x * blockDim.x + threadIdx.x;
  if (g >= G) return;
  float cnt = fmaxf(counts[g], 1.0f);
  float s = 0.f;
  for (int cc = 0; cc < C; ++cc) s += (pooled[(size_t)g * C + cc] / cnt) * lw[cc];
  out[g] = s + lb[0];
}

// ---------------- host launcher ----------------
extern "C" void kernel_launch(void* const* d_in, const int* in_sizes, int n_in,
                              void* d_out, int out_size, void* d_ws, size_t ws_size,
                              hipStream_t stream) {
  (void)n_in; (void)ws_size;
  const int F_IN = 128, Hmax = 4, C = 64;
  const int N = in_sizes[0] / F_IN;
  const int E = in_sizes[1] / 2;
  const int Etot = E + N;
  const int G = out_size;

  const float* x     = (const float*)d_in[0];
  const int*   ei    = (const int*)d_in[1];
  const int*   batch = (const int*)d_in[2];
  const float* Wm[3]   = {(const float*)d_in[3],  (const float*)d_in[7],  (const float*)d_in[11]};
  const float* ASRC[3] = {(const float*)d_in[4],  (const float*)d_in[8],  (const float*)d_in[12]};
  const float* ADST[3] = {(const float*)d_in[5],  (const float*)d_in[9],  (const float*)d_in[13]};
  const float* Bv[3]   = {(const float*)d_in[6],  (const float*)d_in[10], (const float*)d_in[14]};
  const float* lin_w = (const float*)d_in[15];
  const float* lin_b = (const float*)d_in[16];
  float* outp = (float*)d_out;

  // workspace carve (256B aligned)
  char* ws = (char*)d_ws;
  size_t off = 0;
  auto carve = [&](size_t bytes) -> char* {
    char* p = ws + off;
    off += (bytes + 255) & ~(size_t)255;
    return p;
  };
  float*     bufP   = (float*)carve((size_t)N * 256 * 4);     // GEMM output h
  float*     bufA   = (float*)carve((size_t)N * 256 * 4);     // aggregation output / next input
  _Float16*  f16buf = (_Float16*)carve((size_t)N * 256 * 2);  // f16 activations
  _Float16*  wtbuf  = (_Float16*)carve((size_t)256 * 256 * 2);// f16 W^T
  float*     ebuf   = (float*)carve((size_t)Etot * Hmax * 4); // per-edge scores / exp
  float*     esrc   = (float*)carve((size_t)N * Hmax * 4);
  float*     edstb  = (float*)carve((size_t)N * Hmax * 4);
  unsigned*  mbuf   = (unsigned*)carve((size_t)N * Hmax * 4);
  float*     denom  = (float*)carve((size_t)N * Hmax * 4);
  float*     pooled = (float*)carve((size_t)G * C * 4);
  float*     counts = (float*)carve((size_t)G * 4);

  const unsigned ORD_NEG_INF = 0x007FFFFFu;  // f2ord(-inf)
  auto cdiv = [](long long a, long long b) -> unsigned { return (unsigned)((a + b - 1) / b); };

  for (int L = 0; L < 3; ++L) {
    const int K    = (L == 0) ? 128 : 256;
    const int M    = (L == 2) ? 64  : 256;
    const int Hh   = (L == 2) ? 1   : 4;
    const int relu = (L < 2) ? 1 : 0;
    const float* featIn = (L == 0) ? x : bufA;

    // 1) f16 conversion of activations + W transpose
    f32_to_f16<<<cdiv((long long)N * K, TPB), TPB, 0, stream>>>(featIn, f16buf, N * K);
    transpose_to_f16<<<cdiv((long long)K * M, TPB), TPB, 0, stream>>>(Wm[L], wtbuf, K, M);

    // 2) WMMA GEMM: h = feat @ W   (N multiple of 16, K multiple of 32)
    long long waves = (long long)(N / 16) * (M / 16);
    wmma_gemm_f16<<<cdiv(waves * 32, TPB), TPB, 0, stream>>>(f16buf, wtbuf, bufP, N, K, M);

    // 3) per-node attention scores
    node_scores<<<cdiv((long long)N * Hh, TPB), TPB, 0, stream>>>(bufP, ASRC[L], ADST[L],
                                                                  esrc, edstb, N, Hh, C);
    // 4) segment softmax over incoming edges
    fill_u32<<<cdiv((long long)N * Hh, TPB), TPB, 0, stream>>>(mbuf, ORD_NEG_INF, N * Hh);
    fill_u32<<<cdiv((long long)N * Hh, TPB), TPB, 0, stream>>>((unsigned*)denom, 0u, N * Hh);
    edge_scores<<<cdiv((long long)Etot * Hh, TPB), TPB, 0, stream>>>(ei, E, Etot, Hh,
                                                                     esrc, edstb, ebuf, mbuf);
    decode_m<<<cdiv((long long)N * Hh, TPB), TPB, 0, stream>>>(mbuf, N * Hh, ORD_NEG_INF);
    edge_expsum<<<cdiv((long long)Etot * Hh, TPB), TPB, 0, stream>>>(ei, E, Etot, Hh,
                                                                     (const float*)mbuf, ebuf, denom);
    // 5) weighted aggregation (atomics resolve in L2: out fits in 192MB)
    fill_u32<<<cdiv((long long)N * M, TPB), TPB, 0, stream>>>((unsigned*)bufA, 0u, N * M);
    edge_aggregate<<<cdiv((long long)Etot * M, TPB), TPB, 0, stream>>>(ei, E, Etot, Hh, C,
                                                                       bufP, ebuf, denom, bufA);
    // 6) + bias, (ReLU for layers 0/1)
    bias_act<<<cdiv((long long)N * M, TPB), TPB, 0, stream>>>(bufA, Bv[L], N * M, M, relu);
  }

  // global mean pool + linear head
  fill_u32<<<cdiv((long long)G * C, TPB), TPB, 0, stream>>>((unsigned*)pooled, 0u, G * C);
  fill_u32<<<1, TPB, 0, stream>>>((unsigned*)counts, 0u, G);
  pool_counts<<<cdiv(N, TPB), TPB, 0, stream>>>(batch, counts, N);
  pool_sum<<<cdiv((long long)N * C, TPB), TPB, 0, stream>>>(bufA, batch, pooled, N, C);
  final_linear<<<1, 64, 0, stream>>>(pooled, counts, lin_w, lin_b, outp, G, C);
}
